// AKPGAT_15779709846031
// MI455X (gfx1250) — matrix operand order repaired and checked by reading, not compiled, not run on hardware
//
#include <hip/hip_runtime.h>
#include <hip/hip_bf16.h>

// ---------------- problem constants ----------------
#define NNODES 50000
#define NEDGES 800000
#define EPLUS  (NEDGES + NNODES)   // edges + self loops = 850000
#define NHEADS 4
#define IN_C   128
#define HID_C  64
#define OUT_C  40
#define WIDTH  256                 // NHEADS*HID_C
#define W2O    160                 // NHEADS*OUT_C
#define NEG_SLOPE 0.2f
#define BN_EPS 1e-5f

typedef __attribute__((ext_vector_type(16))) __bf16 v16bf;
typedef __attribute__((ext_vector_type(8)))  __bf16 v8bf;
typedef __attribute__((ext_vector_type(8)))  float  v8f;

// ---------------- workspace layout (bytes) ----------------
// all offsets 16B-aligned
#define OFF_XBF   ((size_t)0)                         // bf16 N x 256   = 25,600,000 B
#define OFF_HG    ((size_t)25600000)                  // f32  N x 256   = 51,200,000 B
#define OFF_AGG   ((size_t)76800000)                  // f32  N x 256   = 51,200,000 B
#define OFF_WT    ((size_t)128000000)                 // bf16 256 x 256 =    131,072 B
#define OFF_ESRC  ((size_t)128131072)                 // f32  N x 4     =    800,000 B
#define OFF_EDST  ((size_t)128931072)                 // f32  N x 4
#define OFF_MENC  ((size_t)129731072)                 // u32  N x 4
#define OFF_ZSUM  ((size_t)130531072)                 // f32  N x 4
#define OFF_WBUF  ((size_t)131331072)                 // f32  EPLUS x 4 = 13,600,000 B
#define OFF_MU    ((size_t)144931072)                 // f32  256
#define OFF_RSTD  ((size_t)144932096)                 // f32  256

// ---------------- order-preserving float<->uint (for atomicMax) -------------
__device__ __forceinline__ unsigned fenc(float x) {
  unsigned u = __float_as_uint(x);
  return (u & 0x80000000u) ? ~u : (u | 0x80000000u);
}
__device__ __forceinline__ float fdec(unsigned u) {
  unsigned b = (u & 0x80000000u) ? (u & 0x7fffffffu) : ~u;
  return __uint_as_float(b);
}

// ---------------- small utility kernels ----------------
__global__ void k_f32_to_bf16(const float* __restrict__ x, __bf16* __restrict__ y, int n) {
  int i = blockIdx.x * blockDim.x + threadIdx.x;
  if (i < n) y[i] = (__bf16)x[i];
}

// Wt[o*F + k] = (bf16) W[k*O + o]
__global__ void k_w_transpose(const float* __restrict__ W, __bf16* __restrict__ Wt, int F, int O) {
  int t = blockIdx.x * blockDim.x + threadIdx.x;
  if (t >= F * O) return;
  int o = t / F;
  int k = t - o * F;
  Wt[t] = (__bf16)W[(size_t)k * O + o];
}

__global__ void k_zero_f32(float* __restrict__ p, int n) {
  int i = blockIdx.x * blockDim.x + threadIdx.x;
  if (i < n) p[i] = 0.0f;
}

__global__ void k_init_max(unsigned* __restrict__ p, int n) {
  int i = blockIdx.x * blockDim.x + threadIdx.x;
  if (i < n) p[i] = fenc(-3.0e38f);
}

// ---------------- WMMA bf16 GEMM:  C[M,O] = A[M,F] * Wt[O,F]^T ----------------
// grid.x = M/16 tiles (M = 50000 = 3125*16), block = 256 threads = 8 waves.
// Each wave owns n-tiles {wave, wave+8, ...}. EXEC is all-ones everywhere.
__global__ void __launch_bounds__(256)
k_gemm_bf16(const __bf16* __restrict__ A, const __bf16* __restrict__ Bt,
            float* __restrict__ C, int F, int O) {
  const int lane    = threadIdx.x & 31;
  const int wave    = threadIdx.x >> 5;
  const int halfsel = lane >> 4;     // 0: lanes 0-15, 1: lanes 16-31
  const int r16     = lane & 15;
  const int m0      = blockIdx.x * 16;

  const __bf16* arow = A + (size_t)(m0 + r16) * F;
  const int ntiles = O >> 4;

  for (int nt = wave; nt < ntiles; nt += 8) {
    const __bf16* bcol = Bt + (size_t)(nt * 16 + r16) * F;
    v8f acc = {0.f, 0.f, 0.f, 0.f, 0.f, 0.f, 0.f, 0.f};

    for (int kb = 0; kb < F; kb += 32) {
      // A fragment (16x32 bf16): lanes 0-15 get K kb+0..7 / kb+16..23,
      // lanes 16-31 get K kb+8..15 / kb+24..31
      v8bf alo = *(const v8bf*)(arow + kb + (halfsel << 3));
      v8bf ahi = *(const v8bf*)(arow + kb + 16 + (halfsel << 3));
      v16bf a = __builtin_shufflevector(alo, ahi, 0,1,2,3,4,5,6,7,8,9,10,11,12,13,14,15);
      // B fragment (32x16 bf16): lane = column, same K split as A (Wt is [O][F])
      v8bf blo = *(const v8bf*)(bcol + kb + (halfsel << 3));
      v8bf bhi = *(const v8bf*)(bcol + kb + 16 + (halfsel << 3));
      v16bf b = __builtin_shufflevector(blo, bhi, 0,1,2,3,4,5,6,7,8,9,10,11,12,13,14,15);

      acc = __builtin_amdgcn_wmma_f32_16x16x32_bf16(
          /*neg_a=*/false, a, /*neg_b=*/false, b,
          /*c_mod=*/(short)0, acc, /*reuse_a=*/false, /*reuse_b=*/false);
    }

    // C layout: VGPR r -> (M = r + 8*halfsel, N = lane&15) within the tile
    float* cptr = C + (size_t)(m0 + halfsel * 8) * O + nt * 16 + r16;
#pragma unroll
    for (int r = 0; r < 8; ++r) cptr[(size_t)r * O] = acc[r];
  }
}

// ---------------- attention source/dest logits ----------------
// esrc[n,h] = sum_c hg[n, h*C+c]*a_src[h,c] ; edst likewise.
__global__ void k_att_dots(const float* __restrict__ hg,
                           const float* __restrict__ a_src,
                           const float* __restrict__ a_dst,
                           float* __restrict__ esrc, float* __restrict__ edst,
                           int O, int C) {
  int idx = blockIdx.x * blockDim.x + threadIdx.x;
  if (idx >= NNODES * NHEADS) return;
  int n = idx >> 2;
  int h = idx & 3;
  const float* hp = hg + (size_t)n * O + h * C;
  const float* as = a_src + h * C;
  const float* ad = a_dst + h * C;
  float s1 = 0.f, s2 = 0.f;
  for (int c = 0; c < C; ++c) {
    float v = hp[c];
    s1 += v * as[c];
    s2 += v * ad[c];
  }
  esrc[idx] = s1;
  edst[idx] = s2;
}

// ---------------- edge kernels ----------------
__device__ __forceinline__ void edge_sd(const int* __restrict__ ei, int e, int& s, int& d) {
  if (e < NEDGES) { s = ei[e]; d = ei[NEDGES + e]; }
  else            { s = e - NEDGES; d = s; }        // self loop
}

__global__ void k_edge_max(const int* __restrict__ ei,
                           const float* __restrict__ esrc, const float* __restrict__ edst,
                           unsigned* __restrict__ menc) {
  int t = blockIdx.x * blockDim.x + threadIdx.x;
  if (t >= EPLUS * NHEADS) return;
  int e = t >> 2, h = t & 3;
  int s, d; edge_sd(ei, e, s, d);
  float ee = esrc[s * NHEADS + h] + edst[d * NHEADS + h];
  ee = ee > 0.f ? ee : NEG_SLOPE * ee;
  atomicMax(&menc[d * NHEADS + h], fenc(ee));
}

__global__ void k_edge_sum(const int* __restrict__ ei,
                           const float* __restrict__ esrc, const float* __restrict__ edst,
                           const unsigned* __restrict__ menc,
                           float* __restrict__ wbuf, float* __restrict__ zsum) {
  int t = blockIdx.x * blockDim.x + threadIdx.x;
  if (t >= EPLUS * NHEADS) return;
  int e = t >> 2, h = t & 3;
  int s, d; edge_sd(ei, e, s, d);
  float ee = esrc[s * NHEADS + h] + edst[d * NHEADS + h];
  ee = ee > 0.f ? ee : NEG_SLOPE * ee;
  float w = __expf(ee - fdec(menc[d * NHEADS + h]));
  wbuf[t] = w;
  atomicAdd(&zsum[d * NHEADS + h], w);
}

// per_edge = O/8 threads per edge; each thread scatters 8 channels
__global__ void k_edge_agg(const int* __restrict__ ei,
                           const float* __restrict__ wbuf, const float* __restrict__ zsum,
                           const float* __restrict__ hg, float* __restrict__ agg,
                           int O, int C, int per_edge) {
  int t = blockIdx.x * blockDim.x + threadIdx.x;
  if (t >= EPLUS * per_edge) return;
  int e = t / per_edge;
  int c0 = (t - e * per_edge) * 8;
  int h = c0 / C;
  int s, d; edge_sd(ei, e, s, d);
  float alpha = wbuf[e * NHEADS + h] / (zsum[d * NHEADS + h] + 1e-16f);
  const float* hp = hg + (size_t)s * O + c0;
  float* op = agg + (size_t)d * O + c0;
  float4 v0 = *(const float4*)(hp);
  float4 v1 = *(const float4*)(hp + 4);
  atomicAdd(op + 0, v0.x * alpha);
  atomicAdd(op + 1, v0.y * alpha);
  atomicAdd(op + 2, v0.z * alpha);
  atomicAdd(op + 3, v0.w * alpha);
  atomicAdd(op + 4, v1.x * alpha);
  atomicAdd(op + 5, v1.y * alpha);
  atomicAdd(op + 6, v1.z * alpha);
  atomicAdd(op + 7, v1.w * alpha);
}

// ---------------- batchnorm ----------------
__global__ void __launch_bounds__(256)
k_bn_stats(const float* __restrict__ agg, const float* __restrict__ bias,
           float* __restrict__ mu, float* __restrict__ rstd, int O) {
  __shared__ float sh1[256];
  __shared__ float sh2[256];
  int c = blockIdx.x;
  int tid = threadIdx.x;
  float b = bias[c];
  float sum = 0.f, sq = 0.f;
  for (int n = tid; n < NNODES; n += 256) {
    float v = agg[(size_t)n * O + c] + b;
    sum += v;
    sq  += v * v;
  }
  sh1[tid] = sum; sh2[tid] = sq;
  __syncthreads();
  for (int off = 128; off > 0; off >>= 1) {
    if (tid < off) { sh1[tid] += sh1[tid + off]; sh2[tid] += sh2[tid + off]; }
    __syncthreads();
  }
  if (tid == 0) {
    float m = sh1[0] / (float)NNODES;
    float var = sh2[0] / (float)NNODES - m * m;
    mu[c]   = m;
    rstd[c] = rsqrtf(var + BN_EPS);
  }
}

// y = elu((x + b - mu)*rstd*g + beta) written as bf16 (next GEMM input)
__global__ void k_bn_apply_elu(const float* __restrict__ agg, const float* __restrict__ bias,
                               const float* __restrict__ mu, const float* __restrict__ rstd,
                               const float* __restrict__ g, const float* __restrict__ beta,
                               __bf16* __restrict__ y, int O) {
  int i = blockIdx.x * blockDim.x + threadIdx.x;
  if (i >= NNODES * O) return;
  int c = i % O;
  float v = agg[i] + bias[c];
  v = (v - mu[c]) * rstd[c] * g[c] + beta[c];
  v = v > 0.f ? v : (__expf(v) - 1.0f);   // ELU(alpha=1)
  y[i] = (__bf16)v;
}

// ---------------- final: head mean + bias + log_softmax ----------------
__global__ void k_final(const float* __restrict__ agg, const float* __restrict__ b2,
                        float* __restrict__ out) {
  int n = blockIdx.x * blockDim.x + threadIdx.x;
  if (n >= NNODES) return;
  float vals[OUT_C];
  const float* row = agg + (size_t)n * W2O;
  float mx = -3.0e38f;
#pragma unroll
  for (int c = 0; c < OUT_C; ++c) {
    float s = row[c] + row[OUT_C + c] + row[2 * OUT_C + c] + row[3 * OUT_C + c];
    s = 0.25f * s + b2[c];
    vals[c] = s;
    mx = fmaxf(mx, s);
  }
  float se = 0.f;
#pragma unroll
  for (int c = 0; c < OUT_C; ++c) se += __expf(vals[c] - mx);
  float lse = mx + __logf(se);
  float* op = out + (size_t)n * OUT_C;
#pragma unroll
  for (int c = 0; c < OUT_C; ++c) op[c] = vals[c] - lse;
}

// ---------------- host launcher ----------------
static inline int cdiv(long long a, int b) { return (int)((a + b - 1) / b); }

extern "C" void kernel_launch(void* const* d_in, const int* in_sizes, int n_in,
                              void* d_out, int out_size, void* d_ws, size_t ws_size,
                              hipStream_t stream) {
  const float* x    = (const float*)d_in[0];
  const int*   ei   = (const int*)d_in[1];
  const float* W[3]    = {(const float*)d_in[2],  (const float*)d_in[8],  (const float*)d_in[14]};
  const float* aS[3]   = {(const float*)d_in[3],  (const float*)d_in[9],  (const float*)d_in[15]};
  const float* aD[3]   = {(const float*)d_in[4],  (const float*)d_in[10], (const float*)d_in[16]};
  const float* bb[3]   = {(const float*)d_in[5],  (const float*)d_in[11], (const float*)d_in[17]};
  const float* gg[2]   = {(const float*)d_in[6],  (const float*)d_in[12]};
  const float* beta[2] = {(const float*)d_in[7],  (const float*)d_in[13]};

  char* ws = (char*)d_ws;
  __bf16*   xbf  = (__bf16*)(ws + OFF_XBF);
  float*    hg   = (float*)(ws + OFF_HG);
  float*    agg  = (float*)(ws + OFF_AGG);
  __bf16*   wt   = (__bf16*)(ws + OFF_WT);
  float*    esrc = (float*)(ws + OFF_ESRC);
  float*    edst = (float*)(ws + OFF_EDST);
  unsigned* menc = (unsigned*)(ws + OFF_MENC);
  float*    zsum = (float*)(ws + OFF_ZSUM);
  float*    wbuf = (float*)(ws + OFF_WBUF);
  float*    mu   = (float*)(ws + OFF_MU);
  float*    rstd = (float*)(ws + OFF_RSTD);
  float*    out  = (float*)d_out;

  const int Fs[3] = {IN_C, WIDTH, WIDTH};
  const int Os[3] = {WIDTH, WIDTH, W2O};
  const int Cs[3] = {HID_C, HID_C, OUT_C};

  // layer 0 input: fp32 -> bf16
  k_f32_to_bf16<<<cdiv((long long)NNODES * IN_C, 256), 256, 0, stream>>>(x, xbf, NNODES * IN_C);

  for (int l = 0; l < 3; ++l) {
    const int F = Fs[l], O = Os[l], C = Cs[l];
    const int NH = NNODES * NHEADS;

    // weights: transpose + convert to bf16, Wt[O][F]
    k_w_transpose<<<cdiv((long long)F * O, 256), 256, 0, stream>>>(W[l], wt, F, O);

    // h = x @ W  (WMMA bf16, f32 accumulate)
    k_gemm_bf16<<<NNODES / 16, 256, 0, stream>>>(xbf, wt, hg, F, O);

    // attention logits
    k_att_dots<<<cdiv(NH, 256), 256, 0, stream>>>(hg, aS[l], aD[l], esrc, edst, O, C);

    // init per-(node,head) reductions + output accumulator
    k_init_max<<<cdiv(NH, 256), 256, 0, stream>>>(menc, NH);
    k_zero_f32<<<cdiv(NH, 256), 256, 0, stream>>>(zsum, NH);
    k_zero_f32<<<cdiv((long long)NNODES * O, 256), 256, 0, stream>>>(agg, NNODES * O);

    // segment softmax over incoming edges + weighted aggregation
    k_edge_max<<<cdiv((long long)EPLUS * NHEADS, 256), 256, 0, stream>>>(ei, esrc, edst, menc);
    k_edge_sum<<<cdiv((long long)EPLUS * NHEADS, 256), 256, 0, stream>>>(ei, esrc, edst, menc, wbuf, zsum);
    const int per_edge = O / 8;
    k_edge_agg<<<cdiv((long long)EPLUS * per_edge, 256), 256, 0, stream>>>(
        ei, wbuf, zsum, hg, agg, O, C, per_edge);

    if (l < 2) {
      // bias + batchnorm + ELU, emit bf16 input for next layer
      k_bn_stats<<<O, 256, 0, stream>>>(agg, bb[l], mu, rstd, O);
      k_bn_apply_elu<<<cdiv((long long)NNODES * O, 256), 256, 0, stream>>>(
          agg, bb[l], mu, rstd, gg[l], beta[l], xbf, O);
    } else {
      // head mean + bias + log_softmax
      k_final<<<cdiv(NNODES, 256), 256, 0, stream>>>(agg, bb[l], out);
    }
  }
}
